// PyramidKVCompressor_1211180777496
// MI455X (gfx1250) — matrix operand order repaired and checked by reading, not compiled, not run on hardware
//
#include <hip/hip_runtime.h>
#include <hip/hip_bf16.h>
#include <stdint.h>

typedef __bf16 bf16;
typedef __attribute__((ext_vector_type(16))) __bf16 v16bf;
typedef __attribute__((ext_vector_type(8)))  float  v8f;

#define NTOK 32768        // B*S
#define HDIM 1024
#define TM   64           // tokens per tile
#define XSTR 1032         // padded LDS row stride in elements (16B-aligned rows)

union Frag { uint4 q[2]; v16bf v; };

__device__ __forceinline__ uint32_t pk2bf(float a, float b) {
  union { bf16 h[2]; uint32_t u; } c;
  c.h[0] = (bf16)a; c.h[1] = (bf16)b;
  return c.u;
}

__device__ __forceinline__ v8f wmma_bf16(const Frag& a, const Frag& b, v8f c) {
  return __builtin_amdgcn_wmma_f32_16x16x32_bf16(false, a.v, false, b.v, (short)0, c, false, false);
}

// GEMM over one 64xK A tile (LDS) against two 16-col B strips (global, [N][K]
// layout, K contiguous). Next-K B fragments are prefetched to hide global
// latency; all 4 A fragments are loaded before the WMMA block so the
// scheduler can use partial dscnt waits.
__device__ __forceinline__ void gemm_pair(const bf16* __restrict__ xsrc,
                                          const bf16* __restrict__ b0,
                                          const bf16* __restrict__ b1,
                                          int K, int nsub, int hi,
                                          v8f (&acc0)[4], v8f (&acc1)[4]) {
  Frag fb0, fb1;
  fb0.q[0] = *(const uint4*)(b0);     fb0.q[1] = *(const uint4*)(b0 + 8);
  fb1.q[0] = *(const uint4*)(b1);     fb1.q[1] = *(const uint4*)(b1 + 8);
  for (int k0 = 0; k0 < K; k0 += 32) {
    const int kn = (k0 + 32 < K) ? (k0 + 32) : 0;   // always-valid prefetch addr
    Frag nb0, nb1;
    nb0.q[0] = *(const uint4*)(b0 + kn); nb0.q[1] = *(const uint4*)(b0 + kn + 8);
    nb1.q[0] = *(const uint4*)(b1 + kn); nb1.q[1] = *(const uint4*)(b1 + kn + 8);
    const int aoff = k0 + hi * 16;
    Frag fa[4];
    #pragma unroll
    for (int m = 0; m < 4; ++m) {
      const bf16* arow = xsrc + (m * 16 + nsub) * XSTR + aoff;
      fa[m].q[0] = *(const uint4*)(arow);
      fa[m].q[1] = *(const uint4*)(arow + 8);
    }
    #pragma unroll
    for (int m = 0; m < 4; ++m) {
      acc0[m] = wmma_bf16(fa[m], fb0, acc0[m]);
      acc1[m] = wmma_bf16(fa[m], fb1, acc1[m]);
    }
    fb0 = nb0; fb1 = nb1;
  }
}

__device__ __forceinline__ void gemm_single(const bf16* __restrict__ xsrc,
                                            const bf16* __restrict__ b0,
                                            int K, int nsub, int hi,
                                            v8f (&acc0)[4]) {
  Frag fb0;
  fb0.q[0] = *(const uint4*)(b0); fb0.q[1] = *(const uint4*)(b0 + 8);
  for (int k0 = 0; k0 < K; k0 += 32) {
    const int kn = (k0 + 32 < K) ? (k0 + 32) : 0;
    Frag nb0;
    nb0.q[0] = *(const uint4*)(b0 + kn); nb0.q[1] = *(const uint4*)(b0 + kn + 8);
    const int aoff = k0 + hi * 16;
    Frag fa[4];
    #pragma unroll
    for (int m = 0; m < 4; ++m) {
      const bf16* arow = xsrc + (m * 16 + nsub) * XSTR + aoff;
      fa[m].q[0] = *(const uint4*)(arow);
      fa[m].q[1] = *(const uint4*)(arow + 8);
    }
    #pragma unroll
    for (int m = 0; m < 4; ++m) acc0[m] = wmma_bf16(fa[m], fb0, acc0[m]);
    fb0 = nb0;
  }
}

// ---------------- weight convert + transpose: out[c*R + r] = bf16(in[r*C + c]) ----------------
__global__ void k_transpose_bf16(const float* __restrict__ in, bf16* __restrict__ out,
                                 int R, int C) {
  int i = blockIdx.x * blockDim.x + threadIdx.x;
  if (i >= R * C) return;
  int r = i / C, c = i - r * C;
  out[(size_t)c * R + r] = (bf16)in[i];
}

__global__ void k_zero_counts(int* counts) {
  if (threadIdx.x < 8) counts[threadIdx.x] = 0;
}

// ---------------- predictor: levels for 64-token tiles ----------------
// All 4 levels' Pw1t are contiguous -> treat as one N=1024 GEMM; each strip
// pair falls entirely inside one level.
__global__ void __launch_bounds__(256) k_predict(
    const float* __restrict__ keys, const float* __restrict__ values,
    const bf16* __restrict__ Pw1t,   // [4*256][1024]  (n-major, k contiguous)
    const float* __restrict__ Pb1,   // [1024]
    const float* __restrict__ Pw2,   // [1024]
    const float* __restrict__ Pb2,   // [4]
    int* __restrict__ lev)           // [2][NTOK]
{
  extern __shared__ __align__(16) char smem[];
  bf16*  xs    = (bf16*)smem;                                       // [64][XSTR]
  float* score = (float*)(smem + (size_t)TM * XSTR * sizeof(bf16)); // [4][64]

  const int tid  = threadIdx.x;
  const int lane = tid & 31, wave = tid >> 5;
  const int nsub = lane & 15, hi = lane >> 4;
  const int tens = blockIdx.y;
  const float* x = tens ? values : keys;
  const size_t base = (size_t)blockIdx.x * TM * HDIM;

  { // convert 64x1024 fp32 tile -> bf16 LDS
    int r = tid >> 2, q = tid & 3;
    const float4* src = (const float4*)(x + base + (size_t)r * HDIM) + q * 64;
    bf16* dst = xs + r * XSTR + q * 256;
    #pragma unroll 4
    for (int i = 0; i < 64; ++i) {
      float4 v = src[i];
      uint2 p; p.x = pk2bf(v.x, v.y); p.y = pk2bf(v.z, v.w);
      *(uint2*)(dst + i * 4) = p;
    }
  }
  score[tid] = 0.f;
  __syncthreads();

  for (int p = wave; p < 32; p += 8) {           // 64 strips total, paired
    const int n0 = (2 * p) * 16 + nsub;
    const int n1 = n0 + 16;
    const int l  = (2 * p) >> 4;                 // level of this pair (uniform)
    v8f acc0[4] = {}, acc1[4] = {};
    gemm_pair(xs, Pw1t + (size_t)n0 * HDIM + hi * 16,
                  Pw1t + (size_t)n1 * HDIM + hi * 16,
              HDIM, nsub, hi, acc0, acc1);
    const float bi0 = Pb1[n0], w20 = Pw2[n0];
    const float bi1 = Pb1[n1], w21 = Pw2[n1];
    #pragma unroll
    for (int m = 0; m < 4; ++m) {
      #pragma unroll
      for (int r = 0; r < 8; ++r) {
        float v0 = acc0[m][r] + bi0; v0 = v0 > 0.f ? v0 : 0.f;
        float v1 = acc1[m][r] + bi1; v1 = v1 > 0.f ? v1 : 0.f;
        float c = v0 * w20 + v1 * w21;            // fold pair before reduction
        c += __shfl_xor(c, 1, 32);
        c += __shfl_xor(c, 2, 32);
        c += __shfl_xor(c, 4, 32);
        c += __shfl_xor(c, 8, 32);
        if (nsub == 0) atomicAdd(&score[l * 64 + m * 16 + r + hi * 8], c);
      }
    }
  }
  __syncthreads();
  if (tid < TM) { // sigmoid is monotonic -> argmax over pre-activation; first-max tiebreak
    float best = score[tid] + Pb2[0];
    int lv = 0;
    for (int l = 1; l < 4; ++l) {
      float s = score[l * 64 + tid] + Pb2[l];
      if (s > best) { best = s; lv = l; }
    }
    lev[(size_t)tens * NTOK + blockIdx.x * TM + tid] = lv;
  }
}

// ---------------- bin tokens by level ----------------
__global__ void k_bin(const int* __restrict__ lev, int* __restrict__ counts,
                      int* __restrict__ idxList) {
  int i = blockIdx.x * blockDim.x + threadIdx.x;   // 0..2*NTOK-1
  int t = i >> 15, tok = i & (NTOK - 1);
  int l = lev[i];
  int pos = atomicAdd(&counts[t * 4 + l], 1);
  idxList[(size_t)(t * 4 + l) * NTOK + pos] = tok;
}

// ---------------- per-level bottleneck MLP over gathered tokens ----------------
struct CompArgs {
  const float* x[2];
  float*       out;
  const bf16*  w1t[4];   // [d][1024]
  const bf16*  w2t[4];   // [1024][d]
  const float* b1[4];
  const float* b2[4];
  const int*   counts;   // [2][4]
  const int*   idxList;  // [2][4][NTOK]
};

__global__ void __launch_bounds__(256) k_compress(CompArgs A) {
  extern __shared__ __align__(16) char smem[];
  bf16* xs = (bf16*)smem;                                    // [64][XSTR]
  bf16* hs = (bf16*)(smem + (size_t)TM * XSTR * 2);          // [64][XSTR]
  int*  rowIdx = (int*)(smem + (size_t)TM * XSTR * 4);       // [64]

  const int l = blockIdx.y, t = blockIdx.z;
  const int count = A.counts[t * 4 + l];
  const int tiles = (count + TM - 1) >> 6;
  if ((int)blockIdx.x >= tiles) return;

  const int tid  = threadIdx.x;
  const int lane = tid & 31, wave = tid >> 5;
  const int nsub = lane & 15, hi = lane >> 4;
  const int* idx = A.idxList + (size_t)(t * 4 + l) * NTOK + blockIdx.x * TM;

  if (tid < TM) {
    int g = blockIdx.x * TM + tid;
    rowIdx[tid] = (g < count) ? idx[tid] : -1;
  }
  __syncthreads();

  { // gather + convert 64 token rows -> bf16 LDS
    int r = tid >> 2, q = tid & 3;
    int tok = rowIdx[r]; if (tok < 0) tok = 0;
    const float4* src = (const float4*)(A.x[t] + (size_t)tok * HDIM) + q * 64;
    bf16* dst = xs + r * XSTR + q * 256;
    #pragma unroll 4
    for (int i = 0; i < 64; ++i) {
      float4 v = src[i];
      uint2 p; p.x = pk2bf(v.x, v.y); p.y = pk2bf(v.z, v.w);
      *(uint2*)(dst + i * 4) = p;
    }
  }
  __syncthreads();

  const int d = HDIM >> l;
  // ---- Phase A: hs = relu(x @ w1 + b1), [64 x d] ----
  {
    const bf16*  w1 = A.w1t[l];
    const float* b1 = A.b1[l];
    const int nStrips = d >> 4;
    if (nStrips >= 16) {
      for (int p = wave; p < (nStrips >> 1); p += 8) {
        const int n0 = (2 * p) * 16 + nsub, n1 = n0 + 16;
        v8f acc0[4] = {}, acc1[4] = {};
        gemm_pair(xs, w1 + (size_t)n0 * HDIM + hi * 16,
                      w1 + (size_t)n1 * HDIM + hi * 16,
                  HDIM, nsub, hi, acc0, acc1);
        const float bv0 = b1[n0], bv1 = b1[n1];
        #pragma unroll
        for (int m = 0; m < 4; ++m) {
          #pragma unroll
          for (int r = 0; r < 8; ++r) {
            const int mrow = (m * 16 + r + hi * 8) * XSTR;
            float v0 = acc0[m][r] + bv0; v0 = v0 > 0.f ? v0 : 0.f;
            float v1 = acc1[m][r] + bv1; v1 = v1 > 0.f ? v1 : 0.f;
            hs[mrow + n0] = (bf16)v0;
            hs[mrow + n1] = (bf16)v1;
          }
        }
      }
    } else { // d == 128: 8 strips, one per wave
      const int n0 = wave * 16 + nsub;
      v8f acc0[4] = {};
      gemm_single(xs, w1 + (size_t)n0 * HDIM + hi * 16, HDIM, nsub, hi, acc0);
      const float bv0 = b1[n0];
      #pragma unroll
      for (int m = 0; m < 4; ++m) {
        #pragma unroll
        for (int r = 0; r < 8; ++r) {
          float v0 = acc0[m][r] + bv0; v0 = v0 > 0.f ? v0 : 0.f;
          hs[(m * 16 + r + hi * 8) * XSTR + n0] = (bf16)v0;
        }
      }
    }
  }
  __syncthreads();

  // ---- Phase B: y = hs @ w2 + b2, scatter to out rows ----
  {
    const bf16*  w2 = A.w2t[l];
    const float* b2 = A.b2[l];
    float* outT = A.out + (size_t)t * NTOK * HDIM;
    for (int p = wave; p < 32; p += 8) {         // 64 strips, paired
      const int n0 = (2 * p) * 16 + nsub, n1 = n0 + 16;
      v8f acc0[4] = {}, acc1[4] = {};
      gemm_pair(hs, w2 + (size_t)n0 * d + hi * 16,
                    w2 + (size_t)n1 * d + hi * 16,
                d, nsub, hi, acc0, acc1);
      const float bv0 = b2[n0], bv1 = b2[n1];
      #pragma unroll
      for (int m = 0; m < 4; ++m) {
        #pragma unroll
        for (int r = 0; r < 8; ++r) {
          const int mrow = m * 16 + r + hi * 8;
          const int tok = rowIdx[mrow];
          if (tok >= 0) {
            float* orow = outT + (size_t)tok * HDIM;
            orow[n0] = acc0[m][r] + bv0;
            orow[n1] = acc1[m][r] + bv1;
          }
        }
      }
    }
  }
}

extern "C" void kernel_launch(void* const* d_in, const int* in_sizes, int n_in,
                              void* d_out, int out_size, void* d_ws, size_t ws_size,
                              hipStream_t stream) {
  (void)in_sizes; (void)n_in; (void)out_size; (void)ws_size;

  const float* keys   = (const float*)d_in[0];
  const float* values = (const float*)d_in[1];
  const float* Pw1    = (const float*)d_in[2];
  const float* Pb1    = (const float*)d_in[3];
  const float* Pw2    = (const float*)d_in[4];
  const float* Pb2    = (const float*)d_in[5];

  // workspace layout (all 256B aligned)
  char* ws = (char*)d_ws;
  bf16* Pw1t = (bf16*)ws;                       size_t off = (size_t)4 * 256 * 1024 * 2;
  bf16* cw1t = (bf16*)(ws + off);               off += (size_t)1920 * 1024 * 2;
  bf16* cw2t = (bf16*)(ws + off);               off += (size_t)1920 * 1024 * 2;
  int*  lev     = (int*)(ws + off);             off += (size_t)2 * NTOK * 4;
  int*  counts  = (int*)(ws + off);             off += 256;
  int*  idxList = (int*)(ws + off);

  static const size_t cum[4] = {0, 1048576, 1572864, 1835008}; // elem offsets of per-level w slabs

  for (int l = 0; l < 4; ++l) {
    int d = HDIM >> l;
    int nP = 1024 * 256;
    k_transpose_bf16<<<(nP + 255) / 256, 256, 0, stream>>>(
        Pw1 + (size_t)l * nP, Pw1t + (size_t)l * nP, 1024, 256);
    const float* cw1 = (const float*)d_in[6 + 4 * l];
    const float* cw2 = (const float*)d_in[8 + 4 * l];
    int n1 = 1024 * d;
    k_transpose_bf16<<<(n1 + 255) / 256, 256, 0, stream>>>(cw1, cw1t + cum[l], 1024, d);
    k_transpose_bf16<<<(n1 + 255) / 256, 256, 0, stream>>>(cw2, cw2t + cum[l], d, 1024);
  }

  k_zero_counts<<<1, 32, 0, stream>>>(counts);

  size_t predShm = (size_t)TM * XSTR * 2 + 4 * 64 * 4;   // 133,120 B
  k_predict<<<dim3(NTOK / TM, 2, 1), 256, predShm, stream>>>(
      keys, values, Pw1t, Pb1, Pw2, Pb2, lev);

  k_bin<<<(2 * NTOK) / 256, 256, 0, stream>>>(lev, counts, idxList);

  CompArgs A;
  A.x[0] = keys; A.x[1] = values; A.out = (float*)d_out;
  for (int l = 0; l < 4; ++l) {
    A.w1t[l] = cw1t + cum[l];
    A.w2t[l] = cw2t + cum[l];
    A.b1[l]  = (const float*)d_in[7 + 4 * l];
    A.b2[l]  = (const float*)d_in[9 + 4 * l];
  }
  A.counts = counts; A.idxList = idxList;

  size_t compShm = (size_t)TM * XSTR * 2 * 2 + TM * 4;   // 264,448 B (<= 320KB WGP LDS)
  k_compress<<<dim3(NTOK / TM, 4, 2), 256, compShm, stream>>>(A);
}
